// HierarchicalGNNBlock_23974507446586
// MI455X (gfx1250) — compile-verified, matrix operand order checked by reading, not run on hardware
//
#include <hip/hip_runtime.h>
#include <hip/hip_fp16.h>
#include <math.h>

// Problem constants (from reference)
#define N_NODES  200000
#define N_EDGES  500000
#define N_SNODES 2000
#define N_B      400000
#define N_SE     50000
#define L_ITERS  2

typedef __attribute__((ext_vector_type(16))) _Float16 v16h;
typedef __attribute__((ext_vector_type(8)))  float    v8f;

__device__ __forceinline__ float geluf(float x) {
  // exact GELU (erf), matches jax.nn.gelu(approximate=False)
  return 0.5f * x * (1.0f + erff(x * 0.70710678118654752440f));
}

// ---- WMMA fragment helpers (CDNA5 v_wmma_f32_16x16x32_f16 layouts) ----
// A (16xK f16) from LDS: lane m = lane&15 holds row m; half h -> K = kb + g*8 + (h>=8?16:0) + (h&7)
__device__ __forceinline__ v16h load_a_frag_lds(const _Float16* A, int lda, int kbase, int lane) {
  int m = lane & 15;
  int g = lane >> 4;
  v16h a;
#pragma unroll
  for (int h = 0; h < 16; ++h) {
    int k = kbase + g * 8 + ((h >> 3) << 4) + (h & 7);
    a[h] = A[m * lda + k];
  }
  return a;
}
// B fragment from pre-swizzled f16 weights: fragment f holds 32 lanes x 16 halves contiguously.
__device__ __forceinline__ v16h load_b_frag_sw(const _Float16* __restrict__ wsw, int kb,
                                               int ncol, int ncols, int lane) {
  int f = (kb >> 5) * (ncols >> 4) + (ncol >> 4);
  const _Float16* p = wsw + (size_t)f * 512 + lane * 16;  // 32B aligned per lane
  return *(const v16h*)p;
}

// -------------------- weight pre-swizzle --------------------
// fp32 row-major W[K][Ncols] -> f16 fragment layout: frag f=(kb/32)*(Ncols/16)+(ncol/16),
// element (lane l, half h) = W[kb + (l>>4)*8 + ((h>>3)<<4) + (h&7)][ncol + (l&15)]
__global__ void swz_kernel(const float* __restrict__ w, _Float16* __restrict__ out, int K, int Ncols) {
  int t = blockIdx.x * blockDim.x + threadIdx.x;
  if (t >= K * Ncols) return;
  int f = t >> 9;
  int rem = t & 511;
  int l = rem >> 4, h = rem & 15;
  int npk = Ncols >> 4;
  int kb = (f / npk) << 5;
  int ncol = (f % npk) << 4;
  int k = kb + ((l >> 4) << 3) + ((h >> 3) << 4) + (h & 7);
  int n = ncol + (l & 15);
  out[t] = (_Float16)w[(size_t)k * Ncols + n];
}

// -------------------- utility kernels --------------------
__global__ void zero_kernel(float* __restrict__ p, int n) {
  for (int i = blockIdx.x * blockDim.x + threadIdx.x; i < n; i += gridDim.x * blockDim.x)
    p[i] = 0.0f;
}
__global__ void copy_kernel(float* __restrict__ dst, const float* __restrict__ src, int n) {
  for (int i = blockIdx.x * blockDim.x + threadIdx.x; i < n; i += gridDim.x * blockDim.x)
    dst[i] = src[i];
}
__global__ void invnorm_kernel(const float* __restrict__ x, float* __restrict__ invn, int n) {
  int i = blockIdx.x * blockDim.x + threadIdx.x;
  if (i >= n) return;
  const float* r = x + (size_t)i * 64;
  float s = 0.f;
#pragma unroll
  for (int d = 0; d < 64; ++d) { float v = r[d]; s += v * v; }
  invn[i] = 1.0f / fmaxf(sqrtf(s), 1e-12f);
}

// generic weighted segment-sum scatter: dst[didx[m]][d] += src[(gidx?gidx[m]:m)][d] * rowscale? * wts?
__global__ void scatter_kernel(const float* __restrict__ src,
                               const int* __restrict__ gidx,
                               const int* __restrict__ didx,
                               const float* __restrict__ wts,
                               const float* __restrict__ rowscale,
                               float* __restrict__ dst, int M) {
  int t = blockIdx.x * blockDim.x + threadIdx.x;
  if (t >= M * 64) return;
  int m = t >> 6, d = t & 63;
  int sr = gidx ? gidx[m] : m;
  float v = src[(size_t)sr * 64 + d];
  if (rowscale) v *= rowscale[sr];
  if (wts) v *= wts[m];
  atomicAdd(&dst[(size_t)didx[m] * 64 + d], v);
}

// -------------------- fused gather + 2-layer MLP (WMMA, 32-row tiles) --------------------
// Block = 256 threads = 8 waves; block owns 32 rows (2 M-tiles; B fragments amortized over both).
// y = GELU(x@W1 + b1) @ W2 + b2  [+ GELU] [+ resid]
template <int NSRC>
__global__ __launch_bounds__(256) void mlp64_kernel(
    const float* __restrict__ s0, const int* __restrict__ i0,
    const float* __restrict__ s1, const int* __restrict__ i1,
    const float* __restrict__ s2, const int* __restrict__ i2,
    const _Float16* __restrict__ w1s, const float* __restrict__ b1,  // swz [KIN x 128], [128]
    const _Float16* __restrict__ w2s, const float* __restrict__ b2,  // swz [128 x 64],  [64]
    const float* __restrict__ resid, float* __restrict__ out,        // [M x 64]
    int M, int gelu_out) {
  constexpr int KIN = NSRC * 64;
  __shared__ _Float16 Ain[32][KIN];
  __shared__ _Float16 Hid[32][128];

  const int tid  = threadIdx.x;
  const int lane = tid & 31;
  const int wid  = tid >> 5;
  const int row0 = blockIdx.x * 32;

  // gather input tile -> f16 LDS
  for (int e = tid; e < 32 * KIN; e += 256) {
    int r = e / KIN, k = e % KIN;
    int gr = row0 + r;
    float v = 0.f;
    if (gr < M) {
      int sidx = k >> 6, kk = k & 63;
      const float* sp = (sidx == 0) ? s0 : ((sidx == 1) ? s1 : s2);
      const int*   ip = (sidx == 0) ? i0 : ((sidx == 1) ? i1 : i2);
      int sr = ip ? ip[gr] : gr;
      v = sp[(size_t)sr * 64 + kk];
    }
    Ain[r][k] = (_Float16)v;
  }
  __syncthreads();

  const int g = lane >> 4;
  const int n = lane & 15;

  // GEMM1: [32 x KIN] @ [KIN x 128]; wave wid owns hidden cols [16*wid,16*wid+16), both row tiles
  v8f acc0 = {}, acc1 = {};
  const int ncol = wid * 16;
#pragma unroll
  for (int kb = 0; kb < KIN; kb += 32) {
    v16h b  = load_b_frag_sw(w1s, kb, ncol, 128, lane);
    v16h a0 = load_a_frag_lds(&Ain[0][0],  KIN, kb, lane);
    v16h a1 = load_a_frag_lds(&Ain[16][0], KIN, kb, lane);
    acc0 = __builtin_amdgcn_wmma_f32_16x16x32_f16(false, a0, false, b, (short)0, acc0, false, false);
    acc1 = __builtin_amdgcn_wmma_f32_16x16x32_f16(false, a1, false, b, (short)0, acc1, false, false);
  }
  {
    float bias = b1[ncol + n];
#pragma unroll
    for (int r = 0; r < 8; ++r) {
      Hid[r + 8 * g][ncol + n]      = (_Float16)geluf(acc0[r] + bias);
      Hid[16 + r + 8 * g][ncol + n] = (_Float16)geluf(acc1[r] + bias);
    }
  }
  __syncthreads();

  // GEMM2: [32 x 128] @ [128 x 64]; wave wid -> row tile wid>>2, cols 16*(wid&3)
  {
    const int rt = wid >> 2;
    const int ncol2 = (wid & 3) * 16;
    v8f acc2 = {};
#pragma unroll
    for (int kb = 0; kb < 128; kb += 32) {
      v16h a = load_a_frag_lds(&Hid[rt * 16][0], 128, kb, lane);
      v16h b = load_b_frag_sw(w2s, kb, ncol2, 64, lane);
      acc2 = __builtin_amdgcn_wmma_f32_16x16x32_f16(false, a, false, b, (short)0, acc2, false, false);
    }
    float bias = b2[ncol2 + n];
#pragma unroll
    for (int r = 0; r < 8; ++r) {
      int gr = row0 + rt * 16 + r + 8 * g;
      if (gr < M) {
        float v = acc2[r] + bias;
        if (gelu_out) v = geluf(v);
        if (resid) v += resid[(size_t)gr * 64 + ncol2 + n];
        out[(size_t)gr * 64 + ncol2 + n] = v;
      }
    }
  }
}

// -------------------- BatchNorm statistics --------------------
__global__ void bn_stats_kernel(const float* __restrict__ nodes, const float* __restrict__ snodes,
                                const int* __restrict__ bg0, const int* __restrict__ bg1,
                                float* __restrict__ sums, float* __restrict__ sqs, int M) {
  int c = threadIdx.x;  // 128 columns
  float s = 0.f, q = 0.f;
  for (int r = blockIdx.x; r < M; r += gridDim.x) {
    float v = (c < 64) ? nodes[(size_t)bg0[r] * 64 + c]
                       : snodes[(size_t)bg1[r] * 64 + (c - 64)];
    s += v;
    q += v * v;
  }
  atomicAdd(&sums[c], s);
  atomicAdd(&sqs[c], q);
}
__global__ void bn_finalize_kernel(const float* __restrict__ sums, const float* __restrict__ sqs,
                                   const float* __restrict__ gamma, const float* __restrict__ beta,
                                   float* __restrict__ scale, float* __restrict__ shift, int M) {
  int c = threadIdx.x;
  float mean = sums[c] / (float)M;
  float var  = sqs[c] / (float)M - mean * mean;  // biased, matches jnp.var
  float inv  = rsqrtf(var + 1e-5f);
  float sc   = gamma[c] * inv;
  scale[c] = sc;
  shift[c] = beta[c] - mean * sc;
}

// -------------------- fused readout classifier (32-row tiles) --------------------
// row = BN(concat(nodes[bg0], snodes[bg1])) ; h = GELU(GELU(row@W1+b1)@W2+b2) ; out = h@W3 + b3
__global__ __launch_bounds__(256) void classifier_kernel(
    const float* __restrict__ nodes, const float* __restrict__ snodes,
    const int* __restrict__ bg0, const int* __restrict__ bg1,
    const float* __restrict__ bnscale, const float* __restrict__ bnshift,
    const _Float16* __restrict__ w1s, const float* __restrict__ b1,
    const _Float16* __restrict__ w2s, const float* __restrict__ b2,
    const float* __restrict__ w3, const float* __restrict__ b3,
    float* __restrict__ out, int M) {
  __shared__ _Float16 Ain[32][128];
  __shared__ _Float16 Hid[32][128];
  __shared__ float    H2[32][128];

  const int tid  = threadIdx.x;
  const int lane = tid & 31;
  const int wid  = tid >> 5;
  const int row0 = blockIdx.x * 32;

  for (int e = tid; e < 32 * 128; e += 256) {
    int r = e >> 7, c = e & 127;
    int gr = row0 + r;
    float v = 0.f;
    if (gr < M) {
      float raw = (c < 64) ? nodes[(size_t)bg0[gr] * 64 + c]
                           : snodes[(size_t)bg1[gr] * 64 + (c - 64)];
      v = raw * bnscale[c] + bnshift[c];
    }
    Ain[r][c] = (_Float16)v;
  }
  __syncthreads();

  const int g = lane >> 4;
  const int n = lane & 15;
  const int ncol = wid * 16;

  // layer 1: 128 -> 128, GELU
  {
    v8f acc0 = {}, acc1 = {};
#pragma unroll
    for (int kb = 0; kb < 128; kb += 32) {
      v16h b  = load_b_frag_sw(w1s, kb, ncol, 128, lane);
      v16h a0 = load_a_frag_lds(&Ain[0][0],  128, kb, lane);
      v16h a1 = load_a_frag_lds(&Ain[16][0], 128, kb, lane);
      acc0 = __builtin_amdgcn_wmma_f32_16x16x32_f16(false, a0, false, b, (short)0, acc0, false, false);
      acc1 = __builtin_amdgcn_wmma_f32_16x16x32_f16(false, a1, false, b, (short)0, acc1, false, false);
    }
    float bias = b1[ncol + n];
#pragma unroll
    for (int r = 0; r < 8; ++r) {
      Hid[r + 8 * g][ncol + n]      = (_Float16)geluf(acc0[r] + bias);
      Hid[16 + r + 8 * g][ncol + n] = (_Float16)geluf(acc1[r] + bias);
    }
  }
  __syncthreads();

  // layer 2: 128 -> 128, GELU -> H2 (fp32)
  {
    v8f acc0 = {}, acc1 = {};
#pragma unroll
    for (int kb = 0; kb < 128; kb += 32) {
      v16h b  = load_b_frag_sw(w2s, kb, ncol, 128, lane);
      v16h a0 = load_a_frag_lds(&Hid[0][0],  128, kb, lane);
      v16h a1 = load_a_frag_lds(&Hid[16][0], 128, kb, lane);
      acc0 = __builtin_amdgcn_wmma_f32_16x16x32_f16(false, a0, false, b, (short)0, acc0, false, false);
      acc1 = __builtin_amdgcn_wmma_f32_16x16x32_f16(false, a1, false, b, (short)0, acc1, false, false);
    }
    float bias = b2[ncol + n];
#pragma unroll
    for (int r = 0; r < 8; ++r) {
      H2[r + 8 * g][ncol + n]      = geluf(acc0[r] + bias);
      H2[16 + r + 8 * g][ncol + n] = geluf(acc1[r] + bias);
    }
  }
  __syncthreads();

  // layer 3: 128 -> 1
  if (tid < 32) {
    int gr = row0 + tid;
    if (gr < M) {
      float s = b3[0];
#pragma unroll 8
      for (int c = 0; c < 128; ++c) s += H2[tid][c] * w3[c];
      out[gr] = s;
    }
  }
}

// -------------------- host orchestration --------------------
extern "C" void kernel_launch(void* const* d_in, const int* in_sizes, int n_in,
                              void* d_out, int out_size, void* d_ws, size_t ws_size,
                              hipStream_t stream) {
  (void)in_sizes; (void)n_in; (void)out_size; (void)ws_size;

  // d_in flattening: setup_inputs() dict order; params dict flattened in insertion order.
  const float* nodes_in = (const float*)d_in[0];
  const float* edges_in = (const float*)d_in[1];
  // d_in[2] = semb (unused by reference forward)
  const float* bw = (const float*)d_in[3];   // [B]
  const float* sw = (const float*)d_in[4];   // [SE]
  const float* snode_w1 = (const float*)d_in[5];
  const float* snode_b1 = (const float*)d_in[6];
  const float* snode_w2 = (const float*)d_in[7];
  const float* snode_b2 = (const float*)d_in[8];
  const float* sedge_w1 = (const float*)d_in[9];
  const float* sedge_b1 = (const float*)d_in[10];
  const float* sedge_w2 = (const float*)d_in[11];
  const float* sedge_b2 = (const float*)d_in[12];
  const float* cls_w1 = (const float*)d_in[13];
  const float* cls_b1 = (const float*)d_in[14];
  const float* cls_w2 = (const float*)d_in[15];
  const float* cls_b2 = (const float*)d_in[16];
  const float* cls_w3 = (const float*)d_in[17];
  const float* cls_b3 = (const float*)d_in[18];
  const float* bn_gamma = (const float*)d_in[19];
  const float* bn_beta  = (const float*)d_in[20];
  const float* edge_w1 = (const float*)d_in[21];
  const float* edge_b1 = (const float*)d_in[22];
  const float* edge_w2 = (const float*)d_in[23];
  const float* edge_b2 = (const float*)d_in[24];
  const float* sedgec_w1 = (const float*)d_in[25];
  const float* sedgec_b1 = (const float*)d_in[26];
  const float* sedgec_w2 = (const float*)d_in[27];
  const float* sedgec_b2 = (const float*)d_in[28];
  const float* node_w1 = (const float*)d_in[29];
  const float* node_b1 = (const float*)d_in[30];
  const float* node_w2 = (const float*)d_in[31];
  const float* node_b2 = (const float*)d_in[32];
  const float* snodec_w1 = (const float*)d_in[33];
  const float* snodec_b1 = (const float*)d_in[34];
  const float* snodec_w2 = (const float*)d_in[35];
  const float* snodec_b2 = (const float*)d_in[36];
  const int* graph  = (const int*)d_in[37];  // [2,E]
  const int* bg0    = (const int*)d_in[38];  // [B]
  const int* bg1    = (const int*)d_in[39];  // [B]
  const int* sgraph = (const int*)d_in[40];  // [2,SE]
  const int* g0 = graph;
  const int* g1 = graph + N_EDGES;
  const int* ss0 = sgraph;
  const int* ss1 = sgraph + N_SE;

  // -------- workspace carve-up (~298 MB) --------
  float* w = (float*)d_ws;
  float* nodes_ws = w; w += (size_t)N_NODES * 64;
  float* edges_ws = w; w += (size_t)N_EDGES * 64;
  float* snodes   = w; w += (size_t)N_SNODES * 64;
  float* sedges   = w; w += (size_t)N_SE * 64;
  float* eagg     = w; w += (size_t)N_NODES * 64;
  float* s2n      = w; w += (size_t)N_NODES * 64;
  float* seagg    = w; w += (size_t)N_SNODES * 64;
  float* n2s      = w; w += (size_t)N_SNODES * 64;
  float* invn     = w; w += (size_t)N_NODES;
  float* bnsum    = w; w += 128;
  float* bnsq     = w; w += 128;
  float* bnscale  = w; w += 128;
  float* bnshift  = w; w += 128;
  // swizzled f16 weights
  _Float16* hw = (_Float16*)w;
  auto carve = [&](size_t nh) { _Float16* p = hw; hw += nh; return p; };
  _Float16* snode_w1s = carve(64 * 128);
  _Float16* snode_w2s = carve(128 * 64);
  _Float16* sedge_w1s = carve(128 * 128);
  _Float16* sedge_w2s = carve(128 * 64);
  _Float16* cls_w1s   = carve(128 * 128);
  _Float16* cls_w2s   = carve(128 * 128);
  _Float16* edge_w1s[L_ITERS];   _Float16* edge_w2s[L_ITERS];
  _Float16* sedgec_w1s[L_ITERS]; _Float16* sedgec_w2s[L_ITERS];
  _Float16* node_w1s[L_ITERS];   _Float16* node_w2s[L_ITERS];
  _Float16* snodec_w1s[L_ITERS]; _Float16* snodec_w2s[L_ITERS];
  for (int i = 0; i < L_ITERS; ++i) {
    edge_w1s[i]   = carve(192 * 128); edge_w2s[i]   = carve(128 * 64);
    sedgec_w1s[i] = carve(192 * 128); sedgec_w2s[i] = carve(128 * 64);
    node_w1s[i]   = carve(192 * 128); node_w2s[i]   = carve(128 * 64);
    snodec_w1s[i] = carve(192 * 128); snodec_w2s[i] = carve(128 * 64);
  }

  const int T = 256;
  auto blks = [](long n, int t) { return (unsigned)((n + t - 1) / t); };
  auto swz = [&](const float* src, _Float16* dst, int K, int Nc) {
    swz_kernel<<<blks((long)K * Nc, T), T, 0, stream>>>(src, dst, K, Nc);
  };

  // -------- pre-swizzle all MLP weights to f16 fragment layout --------
  swz(snode_w1, snode_w1s, 64, 128);   swz(snode_w2, snode_w2s, 128, 64);
  swz(sedge_w1, sedge_w1s, 128, 128);  swz(sedge_w2, sedge_w2s, 128, 64);
  swz(cls_w1, cls_w1s, 128, 128);      swz(cls_w2, cls_w2s, 128, 128);
  for (int i = 0; i < L_ITERS; ++i) {
    swz(edge_w1 + (size_t)i * 192 * 128, edge_w1s[i], 192, 128);
    swz(edge_w2 + (size_t)i * 128 * 64,  edge_w2s[i], 128, 64);
    swz(sedgec_w1 + (size_t)i * 192 * 128, sedgec_w1s[i], 192, 128);
    swz(sedgec_w2 + (size_t)i * 128 * 64,  sedgec_w2s[i], 128, 64);
    swz(node_w1 + (size_t)i * 192 * 128, node_w1s[i], 192, 128);
    swz(node_w2 + (size_t)i * 128 * 64,  node_w2s[i], 128, 64);
    swz(snodec_w1 + (size_t)i * 192 * 128, snodec_w1s[i], 192, 128);
    swz(snodec_w2 + (size_t)i * 128 * 64,  snodec_w2s[i], 128, 64);
  }

  // 0) working copies (inputs must not be mutated)
  copy_kernel<<<8192, T, 0, stream>>>(nodes_ws, nodes_in, N_NODES * 64);
  copy_kernel<<<8192, T, 0, stream>>>(edges_ws, edges_in, N_EDGES * 64);
  invnorm_kernel<<<blks(N_NODES, T), T, 0, stream>>>(nodes_ws, invn, N_NODES);

  // 1) snodes = segment_sum(normalize(nodes)[bg0] * bw, bg1)
  zero_kernel<<<blks(N_SNODES * 64, T), T, 0, stream>>>(snodes, N_SNODES * 64);
  scatter_kernel<<<blks((long)N_B * 64, T), T, 0, stream>>>(nodes_ws, bg0, bg1, bw, invn, snodes, N_B);

  // 2) snode encoder (64->128->64, GELU out)
  mlp64_kernel<1><<<blks(N_SNODES, 32), T, 0, stream>>>(
      snodes, nullptr, nullptr, nullptr, nullptr, nullptr,
      snode_w1s, snode_b1, snode_w2s, snode_b2, nullptr, snodes, N_SNODES, 1);

  // 3) sedge encoder (concat(snodes[s0],snodes[s1]) -> 64, GELU out)
  mlp64_kernel<2><<<blks(N_SE, 32), T, 0, stream>>>(
      snodes, ss0, snodes, ss1, nullptr, nullptr,
      sedge_w1s, sedge_b1, sedge_w2s, sedge_b2, nullptr, sedges, N_SE, 1);

  // 4) hierarchical GNN cell iterations
  for (int i = 0; i < L_ITERS; ++i) {
    const size_t ob1 = (size_t)i * 128, ob2 = (size_t)i * 64;

    // edges += MLP(concat(nodes[g0], nodes[g1], edges))
    mlp64_kernel<3><<<blks(N_EDGES, 32), T, 0, stream>>>(
        nodes_ws, g0, nodes_ws, g1, edges_ws, nullptr,
        edge_w1s[i], edge_b1 + ob1, edge_w2s[i], edge_b2 + ob2,
        edges_ws, edges_ws, N_EDGES, 0);

    // sedges += MLP(concat(snodes[s0], snodes[s1], sedges))
    mlp64_kernel<3><<<blks(N_SE, 32), T, 0, stream>>>(
        snodes, ss0, snodes, ss1, sedges, nullptr,
        sedgec_w1s[i], sedgec_b1 + ob1, sedgec_w2s[i], sedgec_b2 + ob2,
        sedges, sedges, N_SE, 0);

    // eagg = segment_sum(edges, g1) ; s2n = segment_sum(snodes[bg1]*bw, bg0)
    zero_kernel<<<8192, T, 0, stream>>>(eagg, N_NODES * 64);
    scatter_kernel<<<blks((long)N_EDGES * 64, T), T, 0, stream>>>(edges_ws, nullptr, g1, nullptr, nullptr, eagg, N_EDGES);
    zero_kernel<<<8192, T, 0, stream>>>(s2n, N_NODES * 64);
    scatter_kernel<<<blks((long)N_B * 64, T), T, 0, stream>>>(snodes, bg1, bg0, bw, nullptr, s2n, N_B);

    // nodes += MLP(concat(eagg, s2n, nodes))
    mlp64_kernel<3><<<blks(N_NODES, 32), T, 0, stream>>>(
        eagg, nullptr, s2n, nullptr, nodes_ws, nullptr,
        node_w1s[i], node_b1 + ob1, node_w2s[i], node_b2 + ob2,
        nodes_ws, nodes_ws, N_NODES, 0);

    // seagg = segment_sum(sedges*sw, s1) ; n2s = segment_sum(nodes[bg0]*bw, bg1)
    zero_kernel<<<blks(N_SNODES * 64, T), T, 0, stream>>>(seagg, N_SNODES * 64);
    scatter_kernel<<<blks((long)N_SE * 64, T), T, 0, stream>>>(sedges, nullptr, ss1, sw, nullptr, seagg, N_SE);
    zero_kernel<<<blks(N_SNODES * 64, T), T, 0, stream>>>(n2s, N_SNODES * 64);
    scatter_kernel<<<blks((long)N_B * 64, T), T, 0, stream>>>(nodes_ws, bg0, bg1, bw, nullptr, n2s, N_B);

    // snodes += MLP(concat(seagg, n2s, snodes))
    mlp64_kernel<3><<<blks(N_SNODES, 32), T, 0, stream>>>(
        seagg, nullptr, n2s, nullptr, snodes, nullptr,
        snodec_w1s[i], snodec_b1 + ob1, snodec_w2s[i], snodec_b2 + ob2,
        snodes, snodes, N_SNODES, 0);
  }

  // 5) BatchNorm statistics over B gathered rows
  zero_kernel<<<1, T, 0, stream>>>(bnsum, 256);  // zeros bnsum+bnsq (contiguous)
  bn_stats_kernel<<<512, 128, 0, stream>>>(nodes_ws, snodes, bg0, bg1, bnsum, bnsq, N_B);
  bn_finalize_kernel<<<1, 128, 0, stream>>>(bnsum, bnsq, bn_gamma, bn_beta, bnscale, bnshift, N_B);

  // 6) classifier readout -> logits
  classifier_kernel<<<blks(N_B, 32), T, 0, stream>>>(
      nodes_ws, snodes, bg0, bg1, bnscale, bnshift,
      cls_w1s, cls_b1, cls_w2s, cls_b2, cls_w3, cls_b3,
      (float*)d_out, N_B);
}
